// Block_32109175505281
// MI455X (gfx1250) — matrix operand order repaired and checked by reading, not compiled, not run on hardware
//
#include <hip/hip_runtime.h>

// ---------------- problem constants ----------------
#define Bb   2
#define Hh   128
#define Ww   256
#define WH   129          // W/2+1
#define Dd   768
#define NBk  8
#define BSz  96
#define HID  3072
#define TOK  (Bb * Hh * Ww)      // 65536 tokens
#define LAMB 0.01f

// ---------------- WMMA types / helpers ----------------
typedef __bf16 v16bf __attribute__((ext_vector_type(16)));
typedef float  v8f   __attribute__((ext_vector_type(8)));

__device__ __forceinline__ v8f wmma_bf16(v16bf a, v16bf b, v8f c) {
    // D = A(16x32 bf16) * B(32x16 bf16) + C(16x16 f32)
    return __builtin_amdgcn_wmma_f32_16x16x32_bf16(false, a, false, b, (short)0, c,
                                                   false, false);
}

// A fragment (16x32): lane = {half = lane>>4, m = lane&15}; row m, K in
// {half*8 .. half*8+7} U {16+half*8 .. 16+half*8+7}  -> two 16B contiguous runs.
__device__ __forceinline__ v16bf load_a_frag(const __bf16* base, int ldm,
                                             int row0, int k0, int lane) {
    int half = lane >> 4, m = lane & 15;
    const __bf16* p = base + (size_t)(row0 + m) * ldm + k0;
    union { uint4 u[2]; v16bf v; } U;
    U.u[0] = *(const uint4*)(p + half * 8);
    U.u[1] = *(const uint4*)(p + 16 + half * 8);
    return U.v;
}

// B fragment (32x16) from K-transposed LDS tile laid out [n][k] with leading dim ldk:
// lane = {half, n}; K in {half*8 .. half*8+7} U {16+half*8 .. 16+half*8+7} (two 16B runs).
__device__ __forceinline__ v16bf load_bt_frag(const __bf16* base, int ldk,
                                              int col0, int k0, int lane) {
    int half = lane >> 4, n = lane & 15;
    const __bf16* p = base + (size_t)(col0 + n) * ldk + k0 + half * 8;
    union { uint4 u[2]; v16bf v; } U;
    U.u[0] = *(const uint4*)(p);
    U.u[1] = *(const uint4*)(p + 16);
    return U.v;
}

__device__ __forceinline__ v16bf neg_frag(v16bf x) {
    v16bf r;
#pragma unroll
    for (int j = 0; j < 16; ++j) r[j] = (__bf16)(-(float)x[j]);
    return r;
}

__device__ __forceinline__ float gelu_exact(float x) {
    return 0.5f * x * (1.0f + erff(x * 0.70710678118654752f));
}

// ---------------- Tensor Data Mover (TDM) 2-D tile load ----------------
#if defined(__gfx1250__) && defined(__has_builtin)
#if __has_builtin(__builtin_amdgcn_tensor_load_to_lds)
#define USE_TDM 1
#endif
#endif

#ifdef USE_TDM
typedef unsigned int v4u __attribute__((ext_vector_type(4)));
typedef int          v4i __attribute__((ext_vector_type(4)));
typedef int          v8i __attribute__((ext_vector_type(8)));

// Load a [tile_d1 rows x tile_d0 elems] tile of 2-byte elements from global
// (row stride = stride0 elements) into LDS at lds_addr. D# per CDNA5 ISA §8.
__device__ __forceinline__ void tdm_load_2d_b16(unsigned lds_addr, const void* gptr,
                                                unsigned tile_d0, unsigned tile_d1,
                                                unsigned tens_d0, unsigned tens_d1,
                                                unsigned long long stride0) {
    unsigned long long ga = (unsigned long long)gptr;
    v4u g0;
    g0[0] = 1u;                                            // count=1, user mode
    g0[1] = lds_addr;                                      // lds_addr [63:32]
    g0[2] = (unsigned)(ga & 0xFFFFFFFFu);                  // global_addr lo
    g0[3] = (unsigned)((ga >> 32) & 0x01FFFFFFu) | 0x80000000u; // ga hi | type=2
    v8i g1;
    g1[0] = (int)(1u << 16);                               // data_size=1 (2 bytes)
    g1[1] = (int)((tens_d0 & 0xFFFFu) << 16);              // tensor_dim0[15:0]@48
    g1[2] = (int)((tens_d0 >> 16) | ((tens_d1 & 0xFFFFu) << 16));
    g1[3] = (int)((tens_d1 >> 16) | (tile_d0 << 16));      // tile_dim0@112
    g1[4] = (int)tile_d1;                                  // tile_dim1@128, tile_dim2=0
    g1[5] = (int)(unsigned)(stride0 & 0xFFFFFFFFu);        // dim0_stride lo
    g1[6] = (int)(unsigned)((stride0 >> 32) & 0xFFFFu);    // dim0_stride hi
    g1[7] = 0;
    v4i z = {0, 0, 0, 0};
#if __clang_major__ >= 23
    v8i z8 = {0, 0, 0, 0, 0, 0, 0, 0};
    __builtin_amdgcn_tensor_load_to_lds(g0, g1, z, z, z8, 0);
#else
    __builtin_amdgcn_tensor_load_to_lds(g0, g1, z, z, 0);
#endif
}
#endif

// ---------------- fp32 -> bf16 conversion ----------------
__global__ void cvt_bf16_kernel(const float* __restrict__ src,
                                __bf16* __restrict__ dst, int n) {
    int i = blockIdx.x * 256 + threadIdx.x;
    if (i < n) dst[i] = (__bf16)src[i];
}

// ---------------- LayerNorm: one wave per 768-ch token ----------------
__global__ __launch_bounds__(256) void ln_kernel(const float* __restrict__ x,
                                                 const float* __restrict__ g,
                                                 const float* __restrict__ b,
                                                 float* __restrict__ outf,
                                                 __bf16* __restrict__ outbf) {
    const int wave = threadIdx.x >> 5, lane = threadIdx.x & 31;
    const int token = blockIdx.x * 8 + wave;
    const float* row = x + (size_t)token * Dd;
    float v[24];
    float s1 = 0.f, s2 = 0.f;
#pragma unroll
    for (int i = 0; i < 24; ++i) {
        float t = row[lane + 32 * i];
        v[i] = t; s1 += t; s2 += t * t;
    }
#pragma unroll
    for (int off = 16; off; off >>= 1) {
        s1 += __shfl_xor(s1, off, 32);
        s2 += __shfl_xor(s2, off, 32);
    }
    const float mean = s1 * (1.0f / Dd);
    const float var  = s2 * (1.0f / Dd) - mean * mean;
    const float rstd = rsqrtf(var + 1e-5f);
#pragma unroll
    for (int i = 0; i < 24; ++i) {
        int c = lane + 32 * i;
        float o = (v[i] - mean) * rstd * g[c] + b[c];
        size_t oi = (size_t)token * Dd + c;
        if (outf) outf[oi] = o;
        else      outbf[oi] = (__bf16)o;
    }
}

// ---------------- LDS radix-2 complex FFT (N/2 threads) ----------------
template <int N>
__device__ __forceinline__ void fft_lds(float* re, float* im, int tid, float dirsign) {
#pragma unroll
    for (int len = 2; len <= N; len <<= 1) {
        __syncthreads();
        int half = len >> 1;
        int grp = tid / half;
        int k = tid - grp * half;
        int i0 = grp * len + k, i1 = i0 + half;
        float ang = dirsign * 6.283185307179586f * (float)k / (float)len;
        float sn, cs;
        __sincosf(ang, &sn, &cs);
        float xr = re[i1], xi = im[i1];
        float tr = cs * xr - sn * xi;
        float ti = cs * xi + sn * xr;
        float ur = re[i0], ui = im[i0];
        re[i0] = ur + tr; im[i0] = ui + ti;
        re[i1] = ur - tr; im[i1] = ui - ti;
    }
    __syncthreads();
}

// forward 256-pt real FFT along W; keep k = 0..128; ortho scale 1/16
__global__ __launch_bounds__(128) void fftw_fwd_kernel(const float* __restrict__ xs,
                                                       float* __restrict__ Fr,
                                                       float* __restrict__ Fi) {
    __shared__ float re[256], im[256];
    const int r = blockIdx.x;
    const int d = r % Dd, h = (r / Dd) % Hh, b = r / (Dd * Hh);
    const int tid = threadIdx.x;
    const float* row = xs + ((size_t)(b * Hh + h) * Ww) * Dd + d;
#pragma unroll
    for (int i = 0; i < 2; ++i) {
        int w = tid + i * 128;
        int p = __brev((unsigned)w) >> 24;
        re[p] = row[(size_t)w * Dd];
        im[p] = 0.f;
    }
    fft_lds<256>(re, im, tid, -1.f);
    const float s = 0.0625f; // 1/sqrt(256)
    size_t obase = ((size_t)(b * Hh + h) * WH) * Dd + d;
    Fr[obase + (size_t)tid * Dd] = re[tid] * s;
    Fi[obase + (size_t)tid * Dd] = im[tid] * s;
    if (tid == 0) {
        Fr[obase + (size_t)128 * Dd] = re[128] * s;
        Fi[obase + (size_t)128 * Dd] = im[128] * s;
    }
}

// 128-pt complex FFT along H (dirsign=-1 forward, +1 inverse); ortho scale 1/sqrt(128)
__global__ __launch_bounds__(64) void ffth_kernel(const float* __restrict__ inR,
                                                  const float* __restrict__ inI,
                                                  float* __restrict__ outR,
                                                  float* __restrict__ outI,
                                                  float dirsign) {
    __shared__ float re[128], im[128];
    const int r = blockIdx.x;
    const int d = r % Dd, k = (r / Dd) % WH, b = r / (Dd * WH);
    const int tid = threadIdx.x;
#pragma unroll
    for (int i = 0; i < 2; ++i) {
        int h = tid + i * 64;
        int p = __brev((unsigned)h) >> 25;
        size_t a = ((size_t)(b * Hh + h) * WH + k) * Dd + d;
        re[p] = inR[a];
        im[p] = inI[a];
    }
    fft_lds<128>(re, im, tid, dirsign);
    const float s = 0.088388347648318447f; // 1/sqrt(128)
#pragma unroll
    for (int i = 0; i < 2; ++i) {
        int h = tid + i * 64;
        size_t a = ((size_t)(b * Hh + h) * WH + k) * Dd + d;
        outR[a] = re[h] * s;
        outI[a] = im[h] * s;
    }
}

// inverse 256-pt FFT along W with Hermitian reconstruction; write real part
__global__ __launch_bounds__(128) void ifftw_kernel(const float* __restrict__ inR,
                                                    const float* __restrict__ inI,
                                                    float* __restrict__ xs) {
    __shared__ float re[256], im[256];
    const int r = blockIdx.x;
    const int d = r % Dd, h = (r / Dd) % Hh, b = r / (Dd * Hh);
    const int tid = threadIdx.x;
    size_t base = ((size_t)(b * Hh + h) * WH) * Dd + d;
#pragma unroll
    for (int i = 0; i < 2; ++i) {
        int w = tid + i * 128;
        float vr, vi;
        if (w <= 128) { vr = inR[base + (size_t)w * Dd]; vi = inI[base + (size_t)w * Dd]; }
        else { vr = inR[base + (size_t)(256 - w) * Dd]; vi = -inI[base + (size_t)(256 - w) * Dd]; }
        int p = __brev((unsigned)w) >> 24;
        re[p] = vr; im[p] = vi;
    }
    fft_lds<256>(re, im, tid, 1.f);
    const float s = 0.0625f;
    float* orow = xs + ((size_t)(b * Hh + h) * Ww) * Dd + d;
#pragma unroll
    for (int i = 0; i < 2; ++i) {
        int w = tid + i * 128;
        orow[(size_t)w * Dd] = re[w] * s;
    }
}

// ---------------- AFNO block-diagonal complex 2-layer MLP via WMMA ----------------
// grid: (P/32, NB); block 256 (8 waves). P = Bb*Hh*WH = 33024 positions.
__global__ __launch_bounds__(256) void mix_kernel(const float* __restrict__ Gr,
                                                  const float* __restrict__ Gi,
                                                  const __bf16* __restrict__ w1bf,
                                                  const __bf16* __restrict__ w2bf,
                                                  const float* __restrict__ b1,
                                                  const float* __restrict__ b2,
                                                  float* __restrict__ Mr,
                                                  float* __restrict__ Mi) {
    __shared__ __align__(16) __bf16 Xr[32][BSz], Xi[32][BSz];
    __shared__ __align__(16) __bf16 Wt[2][BSz][BSz];   // [c][n][m] (K-transposed)
    __shared__ __align__(16) __bf16 Yr[32][BSz], Yi[32][BSz];

    const int tid = threadIdx.x;
    const int lane = tid & 31, wave = tid >> 5;
    const int l = blockIdx.y;
    const int p0 = blockIdx.x * 32;
    const int half = lane >> 4, cn = lane & 15;

    // stage X tile (fp32 -> bf16)
    for (int idx = tid; idx < 32 * BSz; idx += 256) {
        int row = idx / BSz, j = idx - row * BSz;
        size_t a = (size_t)(p0 + row) * Dd + l * BSz + j;
        Xr[row][j] = (__bf16)Gr[a];
        Xi[row][j] = (__bf16)Gi[a];
    }
    // stage layer-1 weights transposed
    for (int idx = tid; idx < 2 * BSz * BSz; idx += 256) {
        int c = idx / (BSz * BSz), rem = idx - c * BSz * BSz;
        int m = rem / BSz, n = rem - m * BSz;
        Wt[c][n][m] = w1bf[((size_t)(c * NBk + l) * BSz + m) * BSz + n];
    }
    __syncthreads();

    // ---- layer 1: R1 = relu(Xr*Wr - Xi*Wi + b1r); I1 = relu(Xr*Wi + Xi*Wr + b1i)
    for (int t = wave; t < 24; t += 8) {
        // tile index is wave-uniform: make it scalar so WMMA runs with full EXEC
        int ts = __builtin_amdgcn_readfirstlane(t);
        int o = ts / 12, tt = ts - o * 12, tr = tt / 6, tc = tt - tr * 6;
        v8f acc = {};
#pragma unroll
        for (int ks = 0; ks < BSz; ks += 32) {
            v16bf ar = load_a_frag(&Xr[0][0], BSz, tr * 16, ks, lane);
            v16bf ai = load_a_frag(&Xi[0][0], BSz, tr * 16, ks, lane);
            v16bf br = load_bt_frag(&Wt[0][0][0], BSz, tc * 16, ks, lane);
            v16bf bi = load_bt_frag(&Wt[1][0][0], BSz, tc * 16, ks, lane);
            if (o == 0) { acc = wmma_bf16(ar, br, acc); acc = wmma_bf16(ai, neg_frag(bi), acc); }
            else        { acc = wmma_bf16(ar, bi, acc); acc = wmma_bf16(ai, br, acc); }
        }
        float bias = b1[(size_t)(o * NBk + l) * BSz + tc * 16 + cn];
#pragma unroll
        for (int rr = 0; rr < 8; ++rr) {
            int row = tr * 16 + half * 8 + rr, col = tc * 16 + cn;
            float v = fmaxf(acc[rr] + bias, 0.f);
            if (o == 0) Yr[row][col] = (__bf16)v;
            else        Yi[row][col] = (__bf16)v;
        }
    }
    __syncthreads();
    // stage layer-2 weights (reuse Wt)
    for (int idx = tid; idx < 2 * BSz * BSz; idx += 256) {
        int c = idx / (BSz * BSz), rem = idx - c * BSz * BSz;
        int m = rem / BSz, n = rem - m * BSz;
        Wt[c][n][m] = w2bf[((size_t)(c * NBk + l) * BSz + m) * BSz + n];
    }
    __syncthreads();

    // ---- layer 2 + softshrink (outputs are >= 0 after relu, so ss == max(v-λ,0))
    for (int t = wave; t < 24; t += 8) {
        int ts = __builtin_amdgcn_readfirstlane(t);
        int o = ts / 12, tt = ts - o * 12, tr = tt / 6, tc = tt - tr * 6;
        v8f acc = {};
#pragma unroll
        for (int ks = 0; ks < BSz; ks += 32) {
            v16bf ar = load_a_frag(&Yr[0][0], BSz, tr * 16, ks, lane);
            v16bf ai = load_a_frag(&Yi[0][0], BSz, tr * 16, ks, lane);
            v16bf br = load_bt_frag(&Wt[0][0][0], BSz, tc * 16, ks, lane);
            v16bf bi = load_bt_frag(&Wt[1][0][0], BSz, tc * 16, ks, lane);
            if (o == 0) { acc = wmma_bf16(ar, br, acc); acc = wmma_bf16(ai, neg_frag(bi), acc); }
            else        { acc = wmma_bf16(ar, bi, acc); acc = wmma_bf16(ai, br, acc); }
        }
        float bias = b2[(size_t)(o * NBk + l) * BSz + tc * 16 + cn];
#pragma unroll
        for (int rr = 0; rr < 8; ++rr) {
            int row = tr * 16 + half * 8 + rr, col = tc * 16 + cn;
            float v = fmaxf(acc[rr] + bias, 0.f);
            v = fmaxf(v - LAMB, 0.f);          // softshrink (v >= 0)
            size_t a = (size_t)(p0 + row) * Dd + l * BSz + col;
            if (o == 0) Mr[a] = v;
            else        Mi[a] = v;
        }
    }
}

// ---------------- bf16 WMMA GEMM: C[M,N] = A[M,K] * B[K,N] (+bias, epilogue) ----
// block tile 128x64, 256 threads = 8 waves in a 4x2 grid, each wave 32x32 (2x2 WMMA)
// A tile is fetched by the Tensor Data Mover (TDM) when available.
__global__ __launch_bounds__(256) void gemm_kernel(const __bf16* __restrict__ A,
                                                   const __bf16* __restrict__ Bw,
                                                   const float* __restrict__ bias,
                                                   int M, int N, int K, int mode,
                                                   __bf16* __restrict__ obf,
                                                   float* __restrict__ of,
                                                   const float* __restrict__ resid) {
    __shared__ __align__(16) __bf16 As[128][32];
    __shared__ __align__(16) __bf16 Bst[64][32];   // [n][k]
    const int tid = threadIdx.x, lane = tid & 31, wave = tid >> 5;
    const int wm = wave >> 1, wn = wave & 1;
    const int bn = blockIdx.x, bm = blockIdx.y;
    const int half = lane >> 4, cn = lane & 15;
    v8f acc[2][2] = {};

    for (int k0 = 0; k0 < K; k0 += 32) {
#ifdef USE_TDM
        // DMA the 128x32 bf16 A tile (row stride K elems) into LDS via TDM
        if (wave == 0) {
            tdm_load_2d_b16((unsigned)(unsigned long long)&As[0][0],
                            A + (size_t)(bm * 128) * K + k0,
                            /*tile_d0=*/32, /*tile_d1=*/128,
                            /*tens_d0=*/(unsigned)(K - k0), /*tens_d1=*/128,
                            /*stride0=*/(unsigned long long)K);
            __builtin_amdgcn_s_wait_tensorcnt(0);
        }
#else
        for (int c = tid; c < 512; c += 256) {
            int off = c * 8, row = off >> 5, col = off & 31;
            *(uint4*)&As[row][col] =
                *(const uint4*)(A + (size_t)(bm * 128 + row) * K + k0 + col);
        }
#endif
        // B tile: 32x64, stored K-transposed [n][k]
        for (int idx = tid; idx < 2048; idx += 256) {
            int k = idx >> 6, n = idx & 63;
            Bst[n][k] = Bw[(size_t)(k0 + k) * N + bn * 64 + n];
        }
        if (k0 + 32 < K)
            __builtin_prefetch(A + (size_t)(bm * 128 + (tid >> 1)) * K + k0 + 32, 0, 1);
        __syncthreads();

        v16bf a0 = load_a_frag(&As[0][0], 32, wm * 32, 0, lane);
        v16bf a1 = load_a_frag(&As[0][0], 32, wm * 32 + 16, 0, lane);
        v16bf b0 = load_bt_frag(&Bst[0][0], 32, wn * 32, 0, lane);
        v16bf b1 = load_bt_frag(&Bst[0][0], 32, wn * 32 + 16, 0, lane);
        acc[0][0] = wmma_bf16(a0, b0, acc[0][0]);
        acc[0][1] = wmma_bf16(a0, b1, acc[0][1]);
        acc[1][0] = wmma_bf16(a1, b0, acc[1][0]);
        acc[1][1] = wmma_bf16(a1, b1, acc[1][1]);
        __syncthreads();
    }

#pragma unroll
    for (int i = 0; i < 2; ++i)
#pragma unroll
        for (int j = 0; j < 2; ++j) {
            int gcol = bn * 64 + wn * 32 + j * 16 + cn;
            float bv = bias[gcol];
#pragma unroll
            for (int rr = 0; rr < 8; ++rr) {
                int grow = bm * 128 + wm * 32 + i * 16 + half * 8 + rr;
                float v = acc[i][j][rr] + bv;
                size_t oidx = (size_t)grow * N + gcol;
                if (mode == 0) obf[oidx] = (__bf16)gelu_exact(v);
                else           of[oidx] = v + resid[oidx];
            }
        }
}

// ---------------- host-side orchestration ----------------
extern "C" void kernel_launch(void* const* d_in, const int* in_sizes, int n_in,
                              void* d_out, int out_size, void* d_ws, size_t ws_size,
                              hipStream_t stream) {
    (void)in_sizes; (void)n_in; (void)out_size; (void)ws_size;
    const float* x    = (const float*)d_in[0];
    const float* n1g  = (const float*)d_in[1];
    const float* n1b  = (const float*)d_in[2];
    const float* w1   = (const float*)d_in[3];
    const float* w2   = (const float*)d_in[4];
    const float* b1   = (const float*)d_in[5];
    const float* b2   = (const float*)d_in[6];
    const float* n2g  = (const float*)d_in[7];
    const float* n2b  = (const float*)d_in[8];
    const float* fc1w = (const float*)d_in[9];
    const float* fc1b = (const float*)d_in[10];
    const float* fc2w = (const float*)d_in[11];
    const float* fc2b = (const float*)d_in[12];
    float* out = (float*)d_out;

    char* ws = (char*)d_ws;
    size_t off = 0;
    auto alloc = [&](size_t bytes) -> char* {
        char* p = ws + off;
        off += (bytes + 255) & ~(size_t)255;
        return p;
    };
    const size_t spatN = (size_t)Bb * Hh * Ww * Dd;   // 50,331,648
    const size_t freqN = (size_t)Bb * Hh * WH * Dd;   // 25,362,432

    float*  Aspat  = (float*)alloc(spatN * 4);
    float*  Fr     = (float*)alloc(freqN * 4);
    float*  Fi     = (float*)alloc(freqN * 4);
    float*  Gr     = (float*)alloc(freqN * 4);
    float*  Gi     = (float*)alloc(freqN * 4);
    __bf16* ln2bf  = (__bf16*)alloc((size_t)TOK * Dd * 2);
    __bf16* h1bf   = (__bf16*)alloc((size_t)TOK * HID * 2);
    __bf16* w1bf   = (__bf16*)alloc((size_t)2 * NBk * BSz * BSz * 2);
    __bf16* w2bf   = (__bf16*)alloc((size_t)2 * NBk * BSz * BSz * 2);
    __bf16* fc1wbf = (__bf16*)alloc((size_t)Dd * HID * 2);
    __bf16* fc2wbf = (__bf16*)alloc((size_t)HID * Dd * 2);

    // weight conversions
    {
        int n = 2 * NBk * BSz * BSz;
        cvt_bf16_kernel<<<(n + 255) / 256, 256, 0, stream>>>(w1, w1bf, n);
        cvt_bf16_kernel<<<(n + 255) / 256, 256, 0, stream>>>(w2, w2bf, n);
        int nf = Dd * HID;
        cvt_bf16_kernel<<<(nf + 255) / 256, 256, 0, stream>>>(fc1w, fc1wbf, nf);
        cvt_bf16_kernel<<<(nf + 255) / 256, 256, 0, stream>>>(fc2w, fc2wbf, nf);
    }

    // LN1 -> fp32 spatial buffer
    ln_kernel<<<TOK / 8, 256, 0, stream>>>(x, n1g, n1b, Aspat, nullptr);

    // forward rfft2
    fftw_fwd_kernel<<<Bb * Hh * Dd, 128, 0, stream>>>(Aspat, Fr, Fi);
    ffth_kernel<<<Bb * WH * Dd, 64, 0, stream>>>(Fr, Fi, Gr, Gi, -1.f);

    // block-diagonal complex MLP + softshrink (WMMA); writes into Fr/Fi
    mix_kernel<<<dim3((Bb * Hh * WH) / 32, NBk), 256, 0, stream>>>(
        Gr, Gi, w1bf, w2bf, b1, b2, Fr, Fi);

    // inverse fft2
    ffth_kernel<<<Bb * WH * Dd, 64, 0, stream>>>(Fr, Fi, Gr, Gi, +1.f);
    ifftw_kernel<<<Bb * Hh * Dd, 128, 0, stream>>>(Gr, Gi, Aspat);

    // LN2 -> bf16
    ln_kernel<<<TOK / 8, 256, 0, stream>>>(Aspat, n2g, n2b, nullptr, ln2bf);

    // MLP: fc1 (+gelu, bf16 out), fc2 (+bias +residual, fp32 out)
    gemm_kernel<<<dim3(HID / 64, TOK / 128), 256, 0, stream>>>(
        ln2bf, fc1wbf, fc1b, TOK, HID, Dd, 0, h1bf, nullptr, nullptr);
    gemm_kernel<<<dim3(Dd / 64, TOK / 128), 256, 0, stream>>>(
        h1bf, fc2wbf, fc2b, TOK, Dd, HID, 1, nullptr, out, x);
}
